// TrueBatchHybridModel_40243843563747
// MI455X (gfx1250) — compile-verified, matrix operand order checked by reading, not run on hardware
//
#include <hip/hip_runtime.h>
#include <hip/hip_bf16.h>

// ---------------- problem constants ----------------
#define Bsz 512
#define Tt  256
#define Nn  500
#define NT  256000        // Bsz*Nn
#define Ee  2048000
#define Hh  128           // LSTM hidden
#define Gg  512           // 4*Hh
#define GHh 64
#define GOo 32

typedef __attribute__((ext_vector_type(16))) __bf16 v16bf;
typedef __attribute__((ext_vector_type(8)))  float  v8f;

union Frag { v16bf v; unsigned u[8]; __bf16 h[16]; };
union Acc  { v8f  v; float f[8]; };

// K index of element pair base for fragment VGPR v, lane-half hi (ISA 16-bit A/B layout)
__device__ __forceinline__ int kmap(int v, int hi) {
    return (v < 4 ? 2 * v : 16 + 2 * (v - 4)) + 8 * hi;
}

// A-fragment (or B-fragment) from an LDS bf16 matrix with row stride S (elements).
__device__ __forceinline__ Frag ldsA(const __bf16* buf, int S, int row, int hi, int kbase) {
    Frag f;
#pragma unroll
    for (int v = 0; v < 8; ++v) {
        int k = kbase + kmap(v, hi);
        f.u[v] = *(const unsigned*)(buf + row * S + k);  // 2 consecutive bf16
    }
    return f;
}

// B-fragment from global f32 weights stored [out=j][in=k] (LSTM Whh/Wih layout)
__device__ __forceinline__ Frag ldB_jk(const float* W, int ldK, int j, int hi, int kbase) {
    Frag f;
#pragma unroll
    for (int v = 0; v < 8; ++v) {
        int k = kbase + kmap(v, hi);
        f.h[2 * v]     = (__bf16)W[j * ldK + k];
        f.h[2 * v + 1] = (__bf16)W[j * ldK + k + 1];
    }
    return f;
}

// B-fragment from global f32 weights stored [in=k][out=j] (MLP layout)
__device__ __forceinline__ Frag ldB_kn(const float* W, int ldN, int j, int hi, int kbase) {
    Frag f;
#pragma unroll
    for (int v = 0; v < 8; ++v) {
        int k = kbase + kmap(v, hi);
        f.h[2 * v]     = (__bf16)W[k * ldN + j];
        f.h[2 * v + 1] = (__bf16)W[(k + 1) * ldN + j];
    }
    return f;
}

__device__ __forceinline__ v8f wmma_bf16(Frag a, Frag b, v8f c) {
    return __builtin_amdgcn_wmma_f32_16x16x32_bf16(false, a.v, false, b.v,
                                                   (short)0, c, false, false);
}

// branchless sigmoid / tanh on v_exp_f32 (saturate correctly, no NaN at +-inf)
__device__ __forceinline__ float sigm(float x) { return 1.0f / (1.0f + __expf(-x)); }
__device__ __forceinline__ float ftanh(float x) {
    float e = __expf(2.0f * x);
    return 1.0f - 2.0f / (e + 1.0f);
}

__device__ __forceinline__ unsigned pack2bf(float a, float b) {
    union { __bf16 h[2]; unsigned u; } p;
    p.h[0] = (__bf16)a; p.h[1] = (__bf16)b;
    return p.u;
}

// low 32 bits of a generic pointer to __shared__ == wave-relative LDS byte address
__device__ __forceinline__ unsigned lds_off(const void* p) {
    return (unsigned)(uintptr_t)p;
}

// =============== LSTM layer 0 ===============
// grid 32 x 1024. Block owns 16 batch rows, runs all 256 steps internally.
// gates = x_t*Wih0^T (rank-1) + h@Whh0^T + bias ; Whh0 B-frags resident in VGPRs.
__global__ __launch_bounds__(1024, 1)
void lstm0_kernel(const float* __restrict__ xseq,   // [B,T,1]
                  const float* __restrict__ Wih0,   // [512,1]
                  const float* __restrict__ Whh0,   // [512,128]
                  const float* __restrict__ bih0,
                  const float* __restrict__ bhh0,
                  __bf16* __restrict__ hs0)         // [T,B,H] bf16
{
    __shared__ __bf16 hb[16 * 130];      // h (bf16), padded rows
    __shared__ float  gat[16 * 513];     // gate tiles f32, padded
    __shared__ float  xall[16 * 256];    // whole input slice for this block (16 KB)
    __shared__ float  biasL[Gg];
    __shared__ float  wih[Gg];

    const int tid  = threadIdx.x;
    const int wave = tid >> 5, lane = tid & 31, ln = lane & 15, hi = lane >> 4;
    const int rowbase = blockIdx.x * 16;
    const int j0 = wave * 16;

    for (int i = tid; i < Gg; i += 1024) { biasL[i] = bih0[i] + bhh0[i]; wih[i] = Wih0[i]; }
    for (int i = tid; i < 16 * 130; i += 1024) hb[i] = (__bf16)0.0f;
    // rows rowbase..rowbase+15 of [B,T] are 4096 contiguous floats
    for (int i = tid; i < 16 * 256; i += 1024) xall[i] = xseq[(size_t)rowbase * Tt + i];

    Frag bw[4];
    {
        int j = j0 + ln;
#pragma unroll
        for (int kb = 0; kb < 4; ++kb) bw[kb] = ldB_jk(Whh0, Hh, j, hi, kb * 32);
    }

    const int ub = tid >> 6, hc = (tid & 63) * 2;   // pointwise mapping: 2 cells/thread
    float c0 = 0.f, c1 = 0.f;

    __syncthreads();
    for (int t = 0; t < Tt; ++t) {
        Acc acc;
#pragma unroll
        for (int i = 0; i < 8; ++i) acc.f[i] = 0.f;
#pragma unroll
        for (int kb = 0; kb < 4; ++kb) {
            Frag a = ldsA(hb, 130, ln, hi, kb * 32);
            acc.v = wmma_bf16(a, bw[kb], acc.v);
        }
#pragma unroll
        for (int v = 0; v < 8; ++v) gat[(v + 8 * hi) * 513 + j0 + ln] = acc.f[v];
        __syncthreads();

        // pointwise LSTM cell update (c kept in registers)
        float xv = xall[ub * 256 + t];
        float h0v, h1v;
        {
            int q = hc;
            float gi = gat[ub * 513 + q]       + biasL[q]       + xv * wih[q];
            float gf = gat[ub * 513 + 128 + q] + biasL[128 + q] + xv * wih[128 + q];
            float gg = gat[ub * 513 + 256 + q] + biasL[256 + q] + xv * wih[256 + q];
            float go = gat[ub * 513 + 384 + q] + biasL[384 + q] + xv * wih[384 + q];
            c0 = sigm(gf) * c0 + sigm(gi) * ftanh(gg);
            h0v = sigm(go) * ftanh(c0);
            q = hc + 1;
            gi = gat[ub * 513 + q]       + biasL[q]       + xv * wih[q];
            gf = gat[ub * 513 + 128 + q] + biasL[128 + q] + xv * wih[128 + q];
            gg = gat[ub * 513 + 256 + q] + biasL[256 + q] + xv * wih[256 + q];
            go = gat[ub * 513 + 384 + q] + biasL[384 + q] + xv * wih[384 + q];
            c1 = sigm(gf) * c1 + sigm(gi) * ftanh(gg);
            h1v = sigm(go) * ftanh(c1);
        }
        unsigned packed = pack2bf(h0v, h1v);
        *(unsigned*)&hb[ub * 130 + hc] = packed;
        *(unsigned*)&hs0[((size_t)t * Bsz + rowbase + ub) * Hh + hc] = packed;
        __syncthreads();
    }
}

// =============== LSTM layer 1 ===============
// gates = hs0_t @ Wih1^T + h1 @ Whh1^T + bias ; both weight B-frag sets in VGPRs.
// hs0 tiles staged one step ahead with async global->LDS (ASYNCcnt), double-buffered.
__global__ __launch_bounds__(1024, 1)
void lstm1_kernel(const __bf16* __restrict__ hs0,  // [T,B,H]
                  const float* __restrict__ Wih1,  // [512,128]
                  const float* __restrict__ Whh1,  // [512,128]
                  const float* __restrict__ bih1,
                  const float* __restrict__ bhh1,
                  float* __restrict__ hlstm)       // [B,128] f32 (final h)
{
    __shared__ __bf16 hb[16 * 130];
    __shared__ __bf16 h0st[2][16 * 130];
    __shared__ float  gat[16 * 513];
    __shared__ float  biasL[Gg];

    const int tid  = threadIdx.x;
    const int wave = tid >> 5, lane = tid & 31, ln = lane & 15, hi = lane >> 4;
    const int rowbase = blockIdx.x * 16;
    const int j0 = wave * 16;

    for (int i = tid; i < Gg; i += 1024) biasL[i] = bih1[i] + bhh1[i];
    for (int i = tid; i < 16 * 130; i += 1024) hb[i] = (__bf16)0.0f;

    Frag bwi[4], bwh[4];
    {
        int j = j0 + ln;
#pragma unroll
        for (int kb = 0; kb < 4; ++kb) {
            bwi[kb] = ldB_jk(Wih1, Hh, j, hi, kb * 32);
            bwh[kb] = ldB_jk(Whh1, Hh, j, hi, kb * 32);
        }
    }

    const int ub = tid >> 6, ul = tid & 63, hc = ul * 2;
    float c0 = 0.f, c1 = 0.f;

    // per-lane addresses for async staging: each lane moves one dword of the tile
    const unsigned ldsA0 = lds_off(&h0st[0][ub * 130 + 2 * ul]);
    const unsigned ldsA1 = lds_off(&h0st[1][ub * 130 + 2 * ul]);
    const __bf16* gsrc = hs0 + ((size_t)rowbase + ub) * Hh + 2 * ul;  // + t*Bsz*Hh per step

    // prologue: stage tile for t=0 into buffer 0
    asm volatile("global_load_async_to_lds_b32 %0, %1, off"
                 :: "v"(ldsA0), "v"(gsrc) : "memory");

    for (int t = 0; t < Tt; ++t) {
        // drain this wave's async transfers, then publish all waves' tiles
        asm volatile("s_wait_asynccnt 0" ::: "memory");
        __syncthreads();   // h0st[cur] complete everywhere; hb from prev step visible

        // prefetch next step's tile into the other buffer (consumed after next barrier)
        if (t + 1 < Tt) {
            unsigned dst = (t & 1) ? ldsA0 : ldsA1;
            const __bf16* src = gsrc + (size_t)(t + 1) * Bsz * Hh;
            asm volatile("global_load_async_to_lds_b32 %0, %1, off"
                         :: "v"(dst), "v"(src) : "memory");
        }

        const __bf16* cur = h0st[t & 1];
        Acc acc;
#pragma unroll
        for (int i = 0; i < 8; ++i) acc.f[i] = 0.f;
#pragma unroll
        for (int kb = 0; kb < 4; ++kb) {
            Frag a = ldsA(cur, 130, ln, hi, kb * 32);
            acc.v = wmma_bf16(a, bwi[kb], acc.v);
        }
#pragma unroll
        for (int kb = 0; kb < 4; ++kb) {
            Frag a = ldsA(hb, 130, ln, hi, kb * 32);
            acc.v = wmma_bf16(a, bwh[kb], acc.v);
        }
#pragma unroll
        for (int v = 0; v < 8; ++v) gat[(v + 8 * hi) * 513 + j0 + ln] = acc.f[v];
        __syncthreads();

        float h0v, h1v;
        {
            int q = hc;
            float gi = gat[ub * 513 + q] + biasL[q];
            float gf = gat[ub * 513 + 128 + q] + biasL[128 + q];
            float gg = gat[ub * 513 + 256 + q] + biasL[256 + q];
            float go = gat[ub * 513 + 384 + q] + biasL[384 + q];
            c0 = sigm(gf) * c0 + sigm(gi) * ftanh(gg);
            h0v = sigm(go) * ftanh(c0);
            q = hc + 1;
            gi = gat[ub * 513 + q] + biasL[q];
            gf = gat[ub * 513 + 128 + q] + biasL[128 + q];
            gg = gat[ub * 513 + 256 + q] + biasL[256 + q];
            go = gat[ub * 513 + 384 + q] + biasL[384 + q];
            c1 = sigm(gf) * c1 + sigm(gi) * ftanh(gg);
            h1v = sigm(go) * ftanh(c1);
        }
        *(unsigned*)&hb[ub * 130 + hc] = pack2bf(h0v, h1v);
        if (t == Tt - 1) {
            hlstm[(size_t)(rowbase + ub) * Hh + hc]     = h0v;
            hlstm[(size_t)(rowbase + ub) * Hh + hc + 1] = h1v;
        }
    }
}

// =============== GCN ===============
__global__ void gcn_init(float* deg, int* tmap) {
    int n = blockIdx.x * blockDim.x + threadIdx.x;
    if (n < NT) { deg[n] = 1.0f; tmap[n] = -1; }   // self-loop weight 1
}

__global__ void gcn_deg(const int* __restrict__ col, const float* __restrict__ w, float* deg) {
    int e = blockIdx.x * blockDim.x + threadIdx.x;
    if (e < Ee) atomicAdd(&deg[col[e]], w[e]);
}

__global__ void gcn_dinv(const float* __restrict__ deg, const float* __restrict__ x,
                         float* dinv, float* s) {
    int n = blockIdx.x * blockDim.x + threadIdx.x;
    if (n < NT) {
        float d = rsqrtf(deg[n]);
        dinv[n] = d;
        s[n] = d * d * x[n];                       // self-loop term of layer-1 scatter
    }
}

__global__ void gcn_scat(const int* __restrict__ row, const int* __restrict__ col,
                         const float* __restrict__ w, const float* __restrict__ x,
                         const float* __restrict__ dinv, float* s) {
    int e = blockIdx.x * blockDim.x + threadIdx.x;
    if (e >= Ee) return;
    int r = row[e], c = col[e];
    atomicAdd(&s[c], dinv[r] * w[e] * dinv[c] * x[r]);
}

// one wave per target: set node->slot map, add self-loop term of layer 2
__global__ void gcn_tgt(const int* __restrict__ tidx, const float* __restrict__ s,
                        const float* __restrict__ dinv,
                        const float* __restrict__ W1, const float* __restrict__ b1,
                        const float* __restrict__ W2,
                        int* tmap, float* acc2) {
    int wv = (blockIdx.x * blockDim.x + threadIdx.x) >> 5;
    int lane = threadIdx.x & 31;
    if (wv >= Bsz) return;
    int n = wv * Nn + tidx[wv];
    if (lane == 0) tmap[n] = wv;
    float d = dinv[n], sn = s[n], a = 0.f;
#pragma unroll 4
    for (int h2 = 0; h2 < GHh; ++h2) {
        float g = fmaxf(sn * W1[h2] + b1[h2], 0.f);   // g1[n][h2] recomputed from s
        a += g * W2[h2 * GOo + lane];
    }
    acc2[wv * GOo + lane] = d * d * a;
}

// only edges landing on a target node (~0.2%) do real work
__global__ void gcn_edges(const int* __restrict__ row, const int* __restrict__ col,
                          const float* __restrict__ w,
                          const float* __restrict__ s, const float* __restrict__ dinv,
                          const int* __restrict__ tmap,
                          const float* __restrict__ W1, const float* __restrict__ b1,
                          const float* __restrict__ W2,
                          float* acc2) {
    int e = blockIdx.x * blockDim.x + threadIdx.x;
    if (e >= Ee) return;
    int c = col[e];
    int slot = tmap[c];
    if (slot < 0) return;
    int r = row[e];
    float nrm = dinv[r] * w[e] * dinv[c];
    float sr = s[r];
    float a[GOo];
#pragma unroll
    for (int j = 0; j < GOo; ++j) a[j] = 0.f;
    for (int h2 = 0; h2 < GHh; ++h2) {
        float g = fmaxf(sr * W1[h2] + b1[h2], 0.f);
#pragma unroll
        for (int j = 0; j < GOo; ++j) a[j] += g * W2[h2 * GOo + j];
    }
#pragma unroll
    for (int j = 0; j < GOo; ++j) atomicAdd(&acc2[slot * GOo + j], nrm * a[j]);
}

// =============== MLP head (fuses GNN relu(+b2), concat, 160->128->64->1) ===============
__global__ __launch_bounds__(256, 1)
void mlp_kernel(const float* __restrict__ hlstm, const float* __restrict__ acc2,
                const float* __restrict__ gb2,
                const float* __restrict__ W0, const float* __restrict__ b0,
                const float* __restrict__ W1, const float* __restrict__ b1,
                const float* __restrict__ W2, const float* __restrict__ b2,
                float* __restrict__ out) {
    __shared__ __bf16 a0[16 * 168];   // K=160 padded
    __shared__ __bf16 a1[16 * 130];
    __shared__ __bf16 a2[16 * 66];

    const int tid  = threadIdx.x;
    const int wave = tid >> 5, lane = tid & 31, ln = lane & 15, hi = lane >> 4;
    const int rowbase = blockIdx.x * 16;

    for (int i = tid; i < 16 * 160; i += 256) {
        int b = i / 160, f = i % 160;
        float v = (f < Hh) ? hlstm[(size_t)(rowbase + b) * Hh + f]
                           : fmaxf(acc2[(rowbase + b) * GOo + (f - Hh)] + gb2[f - Hh], 0.f);
        a0[b * 168 + f] = (__bf16)v;
    }
    Frag w0f[5];
    {
        int j = wave * 16 + ln;
#pragma unroll
        for (int kb = 0; kb < 5; ++kb) w0f[kb] = ldB_kn(W0, 128, j, hi, kb * 32);
    }
    __syncthreads();

    Acc acc;
#pragma unroll
    for (int i = 0; i < 8; ++i) acc.f[i] = 0.f;
#pragma unroll
    for (int kb = 0; kb < 5; ++kb) {
        Frag a = ldsA(a0, 168, ln, hi, kb * 32);
        acc.v = wmma_bf16(a, w0f[kb], acc.v);
    }
    {
        int j = wave * 16 + ln;
        float bb = b0[j];
#pragma unroll
        for (int v = 0; v < 8; ++v)
            a1[(v + 8 * hi) * 130 + j] = (__bf16)fmaxf(acc.f[v] + bb, 0.f);
    }
    __syncthreads();

    if (wave < 4) {
        int j = wave * 16 + ln;
        Frag w1f[4];
#pragma unroll
        for (int kb = 0; kb < 4; ++kb) w1f[kb] = ldB_kn(W1, 64, j, hi, kb * 32);
        Acc acc1;
#pragma unroll
        for (int i = 0; i < 8; ++i) acc1.f[i] = 0.f;
#pragma unroll
        for (int kb = 0; kb < 4; ++kb) {
            Frag a = ldsA(a1, 130, ln, hi, kb * 32);
            acc1.v = wmma_bf16(a, w1f[kb], acc1.v);
        }
        float bb = b1[j];
#pragma unroll
        for (int v = 0; v < 8; ++v)
            a2[(v + 8 * hi) * 66 + j] = (__bf16)fmaxf(acc1.f[v] + bb, 0.f);
    }
    __syncthreads();

    if (tid < 16) {
        float sum = b2[0];
#pragma unroll 4
        for (int h2 = 0; h2 < 64; ++h2) sum += (float)a2[tid * 66 + h2] * W2[h2];
        out[rowbase + tid] = sum;
    }
}

// =============== launch ===============
extern "C" void kernel_launch(void* const* d_in, const int* in_sizes, int n_in,
                              void* d_out, int out_size, void* d_ws, size_t ws_size,
                              hipStream_t stream) {
    const float* xseq  = (const float*)d_in[0];
    const float* x     = (const float*)d_in[1];
    const int*   eidx  = (const int*)d_in[2];      // [2,E]
    const float* ew    = (const float*)d_in[3];
    const int*   tidx  = (const int*)d_in[4];
    const float* Wih0  = (const float*)d_in[5];
    const float* Whh0  = (const float*)d_in[6];
    const float* bih0  = (const float*)d_in[7];
    const float* bhh0  = (const float*)d_in[8];
    const float* Wih1  = (const float*)d_in[9];
    const float* Whh1  = (const float*)d_in[10];
    const float* bih1  = (const float*)d_in[11];
    const float* bhh1  = (const float*)d_in[12];
    const float* gW1   = (const float*)d_in[13];
    const float* gb1   = (const float*)d_in[14];
    const float* gW2   = (const float*)d_in[15];
    const float* gb2   = (const float*)d_in[16];
    const float* mW0   = (const float*)d_in[17];
    const float* mb0   = (const float*)d_in[18];
    const float* mW1   = (const float*)d_in[19];
    const float* mb1   = (const float*)d_in[20];
    const float* mW2   = (const float*)d_in[21];
    const float* mb2   = (const float*)d_in[22];
    float* out = (float*)d_out;

    const int* row = eidx;
    const int* col = eidx + Ee;

    // workspace carving (~38 MB)
    char* ws = (char*)d_ws;
    size_t off = 0;
    __bf16* hs0  = (__bf16*)(ws + off); off += (size_t)Tt * Bsz * Hh * 2;  // 32 MB
    float* hlstm = (float*)(ws + off);  off += (size_t)Bsz * Hh * 4;
    float* deg   = (float*)(ws + off);  off += (size_t)NT * 4;
    float* dinv  = (float*)(ws + off);  off += (size_t)NT * 4;
    float* s     = (float*)(ws + off);  off += (size_t)NT * 4;
    int*   tmap  = (int*)(ws + off);    off += (size_t)NT * 4;
    float* acc2  = (float*)(ws + off);  off += (size_t)Bsz * GOo * 4;
    (void)ws_size; (void)in_sizes; (void)n_in; (void)out_size;

    // LSTM branch (WMMA, time loop inside kernels)
    lstm0_kernel<<<Bsz / 16, 1024, 0, stream>>>(xseq, Wih0, Whh0, bih0, bhh0, hs0);
    lstm1_kernel<<<Bsz / 16, 1024, 0, stream>>>(hs0, Wih1, Whh1, bih1, bhh1, hlstm);

    // GCN branch (bandwidth-minimized)
    gcn_init<<<NT / 256, 256, 0, stream>>>(deg, tmap);
    gcn_deg<<<Ee / 256, 256, 0, stream>>>(col, ew, deg);
    gcn_dinv<<<NT / 256, 256, 0, stream>>>(deg, x, dinv, s);
    gcn_scat<<<Ee / 256, 256, 0, stream>>>(row, col, ew, x, dinv, s);
    gcn_tgt<<<Bsz / 8, 256, 0, stream>>>(tidx, s, dinv, gW1, gb1, gW2, tmap, acc2);
    gcn_edges<<<Ee / 256, 256, 0, stream>>>(row, col, ew, s, dinv, tmap, gW1, gb1, gW2, acc2);

    // fused concat + MLP head (WMMA)
    mlp_kernel<<<Bsz / 16, 256, 0, stream>>>(hlstm, acc2, gb2, mW0, mb0, mW1, mb1, mW2, mb2, out);
}
